// NRMSModel_82197084111646
// MI455X (gfx1250) — matrix operand order repaired and checked by reading, not compiled, not run on hardware
//
#include <hip/hip_runtime.h>

typedef __attribute__((ext_vector_type(16))) _Float16 v16h;
typedef __attribute__((ext_vector_type(8)))  _Float16 h8;
typedef __attribute__((ext_vector_type(8)))  float    v8f;

constexpr int NB    = 256;   // batch
constexpr int NCDD  = 5;     // candidates
constexpr int NHIS  = 50;    // history
constexpr int NT    = 20;    // tokens per title
constexpr int ND    = 300;   // embedding dim
constexpr int NDP   = 320;   // padded embedding dim (10 x 32)
constexpr int NH    = 16;    // heads
constexpr int NVD   = 16;    // value dim per head
constexpr int NREPR = 256;   // H * VD
constexpr int NQD   = 200;   // query dim
constexpr int NQDP  = 208;   // padded (13 x 16)
constexpr float ATT_SCALE = 0.057735026918962574f;  // 1/sqrt(300)

// packed weight geometry: [batch][kc][nt][lane(32)][j(16)] halves,
// value = W[kc*32 + (lane>>4)*16 + j][nt*16 + (lane&15)]
constexpr int PK_WQW_HEAD = 10 * 20 * 512;  // K=300->320, N=300->320
constexpr int PK_WVW_HEAD = 10 * 1 * 512;   // K=300->320, N=16
constexpr int PK_WKW_SZ   = 8 * 13 * 512;   // K=256,      N=200->208
constexpr int PK_WQN_HEAD = 8 * 16 * 512;   // K=256,      N=256
constexpr int PK_WVN_HEAD = 8 * 1 * 512;    // K=256,      N=16
constexpr int PK_WKN_SZ   = 8 * 13 * 512;

// ---------------- WMMA fragment helpers (wave32; A/B use contiguous k = 16*sel + j) -------
// A fragment (16x32 f16) from row-major f16 tile: per lane two contiguous
// 8-half runs: K = sel*16 + [0..7]  and  K = sel*16 + 8 + [0..7]
__device__ __forceinline__ v16h frag_a_lds(const _Float16* p, int ld) {
  const int lane = threadIdx.x & 31, sel = lane >> 4;
  const _Float16* row = p + (lane & 15) * ld + sel * 16;
  h8 lo = *(const h8*)(row);
  h8 hi = *(const h8*)(row + 8);
  v16h a;
#pragma unroll
  for (int j = 0; j < 8; ++j) { a[j] = lo[j]; a[8 + j] = hi[j]; }
  return a;
}

// A fragment from f32 tile with per-element convert (attn rows only)
__device__ __forceinline__ v16h frag_a_f32(const float* p, int ld) {
  const int lane = threadIdx.x & 31, sel = lane >> 4;
  const float* row = p + (lane & 15) * ld;
  v16h a;
#pragma unroll
  for (int v = 0; v < 8; ++v) {
    int k = (sel << 4) + (v << 1);
    a[2 * v]     = (_Float16)row[k];
    a[2 * v + 1] = (_Float16)row[k + 1];
  }
  return a;
}

// B fragment, x^T style: B[k][n] = p[n*ld + k]; per lane contiguous 16 halves
// at K = 16*sel + [0..15]
__device__ __forceinline__ v16h frag_bT_lds(const _Float16* p, int ld) {
  const int lane = threadIdx.x & 31, sel = lane >> 4;
  const _Float16* row = p + (lane & 15) * ld + sel * 16;
  h8 lo = *(const h8*)(row);
  h8 hi = *(const h8*)(row + 8);
  v16h b;
#pragma unroll
  for (int j = 0; j < 8; ++j) { b[j] = lo[j]; b[8 + j] = hi[j]; }
  return b;
}

// B fragment from packed (fragment-ready) storage: 32 contiguous bytes/lane
__device__ __forceinline__ v16h frag_pk(const _Float16* p) {
  return *(const v16h*)(p + (threadIdx.x & 31) * 16);
}

// C/D (16x16 f32): VGPR v -> row M = v + 8*sel, col N = lane&15
__device__ __forceinline__ void frag_c_store(float* p, int ld, v8f d) {
  const int lane = threadIdx.x & 31, sel = lane >> 4, c = lane & 15;
#pragma unroll
  for (int v = 0; v < 8; ++v) p[(v + 8 * sel) * ld + c] = d[v];
}

__device__ __forceinline__ v8f frag_c_load(const float* p, int ld) {
  const int lane = threadIdx.x & 31, sel = lane >> 4, c = lane & 15;
  v8f r;
#pragma unroll
  for (int v = 0; v < 8; ++v) r[v] = p[(v + 8 * sel) * ld + c];
  return r;
}

__device__ __forceinline__ void frag_c_store_f16(_Float16* p, int ld, v8f d) {
  const int lane = threadIdx.x & 31, sel = lane >> 4, c = lane & 15;
#pragma unroll
  for (int v = 0; v < 8; ++v) p[(v + 8 * sel) * ld + c] = (_Float16)d[v];
}

// store xv C-fragment (m-tile mi, rows are future K) into packed-B layout:
// one contiguous h8 per lane
__device__ __forceinline__ void store_xv_pk(_Float16* xvpk, int mi, v8f d) {
  const int lane = threadIdx.x & 31, sel = lane >> 4, c = lane & 15;
  h8 h;
#pragma unroll
  for (int v = 0; v < 8; ++v) h[v] = (_Float16)d[v];
  *(h8*)(xvpk + (mi * 16 + c) * 16 + sel * 8) = h;
}

__device__ __forceinline__ v8f wmma_f16(v16h a, v16h b, v8f c) {
  return __builtin_amdgcn_wmma_f32_16x16x32_f16(false, a, false, b, (short)0, c,
                                                false, false);
}

// ===========================================================================
// Kernel 0: pack an f32 [nb][K][N] weight into f16 B-fragment layout
// ===========================================================================
__global__ void __launch_bounds__(256) nrms_pack_b(
    const float* __restrict__ src, _Float16* __restrict__ dst, int K, int N,
    int ntk, int ntn, int nb) {
  const size_t total = (size_t)nb * ntk * ntn * 512;
  for (size_t idx = (size_t)blockIdx.x * 256 + threadIdx.x; idx < total;
       idx += (size_t)gridDim.x * 256) {
    int j = (int)(idx & 15);
    int lane = (int)((idx >> 4) & 31);
    size_t t = idx >> 9;
    int nt = (int)(t % ntn); t /= ntn;
    int kc = (int)(t % ntk);
    int hb = (int)(t / ntk);
    int k = kc * 32 + (lane >> 4) * 16 + j;
    int n = nt * 16 + (lane & 15);
    float v = (k < K && n < N) ? src[((size_t)hb * K + k) * N + n] : 0.f;
    dst[idx] = (_Float16)v;
  }
}

// ===========================================================================
// Kernel 1: word-level news encoder.  One block per (batch, title).
// ===========================================================================
__global__ void __launch_bounds__(256) nrms_word_encoder(
    const int* __restrict__ cand, const int* __restrict__ clicked,
    const float* __restrict__ emb, const _Float16* __restrict__ pkWq,
    const _Float16* __restrict__ pkWv, const _Float16* __restrict__ pkWk,
    const float* __restrict__ bk, const float* __restrict__ qvec,
    float* __restrict__ cdd_out, float* __restrict__ his_out) {
  extern __shared__ char smem[];
  float*    val  = (float*)smem;                 // [32][256]
  float*    keys = val + 32 * NREPR;             // [32][208]
  float*    sc   = keys + 32 * NQDP;             // [8][32][32] per-wave scores
  _Float16* xs   = (_Float16*)(sc + 8 * 32 * 32);// [32][320]   f16 x
  _Float16* valh = xs + 32 * NDP;                // [32][256]   f16 val
  _Float16* ach  = valh + 32 * NREPR;            // [8][32][32] per-wave A-chunk
  _Float16* xvpk = ach + 8 * 32 * 32;            // [8][32][16] per-wave xv (pk)
  float*    wl   = (float*)(xvpk + 8 * 32 * 16); // [32]
  float*    wp   = wl + 32;                      // [32]
  int*      toks = (int*)(wp + 32);              // [32]

  const int tid = threadIdx.x;
  const int w = tid >> 5, lane = tid & 31;
  const int item = blockIdx.x;
  const int b = item / (NCDD + NHIS);
  const int j = item % (NCDD + NHIS);
  const int* tp;
  float* outp;
  if (j < NCDD) {
    tp = cand + (size_t)(b * NCDD + j) * NT;
    outp = cdd_out + (size_t)(b * NCDD + j) * NREPR;
  } else {
    int jj = j - NCDD;
    tp = clicked + (size_t)(b * NHIS + jj) * NT;
    outp = his_out + (size_t)(b * NHIS + jj) * NREPR;
  }

  if (tid < NT) toks[tid] = tp[tid];
  for (int i = tid; i < 32 * NDP; i += 256) xs[i] = (_Float16)0.f;
  __syncthreads();
  for (int i = tid; i < NT * ND; i += 256) {
    int s = i / ND, d = i - s * ND;
    xs[s * NDP + d] = (_Float16)emb[(size_t)toks[s] * ND + d];
  }
  __syncthreads();

  float*    scw   = sc + w * 32 * 32;
  _Float16* achw  = ach + w * 32 * 32;
  _Float16* xvpkw = xvpk + w * 32 * 16;

  for (int hi = 0; hi < 2; ++hi) {
    const int h = w + hi * 8;
    const _Float16* Wqh = pkWq + (size_t)h * PK_WQW_HEAD;
    const _Float16* Wvh = pkWv + (size_t)h * PK_WVW_HEAD;
    __builtin_prefetch(Wqh, 0, 3);

    // scores = (x @ Wq[h]) @ x^T, fused over 32-wide e-chunks; score acc in regs
    v8f s00 = {}, s01 = {}, s10 = {}, s11 = {};
    for (int ec = 0; ec < 10; ++ec) {
      v8f a00 = {}, a01 = {}, a10 = {}, a11 = {};
      for (int kc = 0; kc < 10; ++kc) {
        v16h fa0 = frag_a_lds(xs + kc * 32, NDP);
        v16h fa1 = frag_a_lds(xs + 16 * NDP + kc * 32, NDP);
        v16h fb0 = frag_pk(Wqh + (size_t)(kc * 20 + 2 * ec) * 512);
        v16h fb1 = frag_pk(Wqh + (size_t)(kc * 20 + 2 * ec + 1) * 512);
        a00 = wmma_f16(fa0, fb0, a00);
        a01 = wmma_f16(fa0, fb1, a01);
        a10 = wmma_f16(fa1, fb0, a10);
        a11 = wmma_f16(fa1, fb1, a11);
      }
      frag_c_store_f16(achw, 32, a00);
      frag_c_store_f16(achw + 16, 32, a01);
      frag_c_store_f16(achw + 16 * 32, 32, a10);
      frag_c_store_f16(achw + 16 * 32 + 16, 32, a11);
      v16h fA0 = frag_a_lds(achw, 32);
      v16h fA1 = frag_a_lds(achw + 16 * 32, 32);
      v16h fT0 = frag_bT_lds(xs + ec * 32, NDP);
      v16h fT1 = frag_bT_lds(xs + 16 * NDP + ec * 32, NDP);
      s00 = wmma_f16(fA0, fT0, s00);
      s01 = wmma_f16(fA0, fT1, s01);
      s10 = wmma_f16(fA1, fT0, s10);
      s11 = wmma_f16(fA1, fT1, s11);
    }
    // dump scores and do the row softmax in f32 LDS
    frag_c_store(scw, 32, s00);
    frag_c_store(scw + 16, 32, s01);
    frag_c_store(scw + 16 * 32, 32, s10);
    frag_c_store(scw + 16 * 32 + 16, 32, s11);
    {
      int s = lane;
      if (s < NT) {
        float m = -1e30f;
        for (int t = 0; t < NT; ++t) m = fmaxf(m, scw[s * 32 + t] * ATT_SCALE);
        float sum = 0.f;
        for (int t = 0; t < NT; ++t) sum += __expf(scw[s * 32 + t] * ATT_SCALE - m);
        float inv = 1.f / sum;
        for (int t = 0; t < NT; ++t)
          scw[s * 32 + t] = __expf(scw[s * 32 + t] * ATT_SCALE - m) * inv;
        for (int t = NT; t < 32; ++t) scw[s * 32 + t] = 0.f;
      } else {
        for (int t = 0; t < 32; ++t) scw[s * 32 + t] = 0.f;
      }
    }
    // xv = x @ Wv[h]  [32 x 16] -> packed-B layout
    {
      v8f x0 = {}, x1 = {};
      for (int kc = 0; kc < 10; ++kc) {
        v16h fa0 = frag_a_lds(xs + kc * 32, NDP);
        v16h fa1 = frag_a_lds(xs + 16 * NDP + kc * 32, NDP);
        v16h fb = frag_pk(Wvh + (size_t)kc * 512);
        x0 = wmma_f16(fa0, fb, x0);
        x1 = wmma_f16(fa1, fb, x1);
      }
      store_xv_pk(xvpkw, 0, x0);
      store_xv_pk(xvpkw, 1, x1);
    }
    // val[:, h*16 : h*16+16] = attn @ xv
    {
      v16h fb = frag_pk(xvpkw);
      v16h fa0 = frag_a_f32(scw, 32);
      v16h fa1 = frag_a_f32(scw + 16 * 32, 32);
      v8f v0 = {}, v1 = {};
      v0 = wmma_f16(fa0, fb, v0);
      v1 = wmma_f16(fa1, fb, v1);
      frag_c_store(val + h * NVD, NREPR, v0);
      frag_c_store(val + 16 * NREPR + h * NVD, NREPR, v1);
      frag_c_store_f16(valh + h * NVD, NREPR, v0);
      frag_c_store_f16(valh + 16 * NREPR + h * NVD, NREPR, v1);
    }
  }
  __syncthreads();

  // key = tanh(val @ Wk + bk)   [32 x 208], N-tiles split across waves
  for (int nt = w; nt < 13; nt += 8) {
    const int c0 = nt * 16;
    v8f k0 = {}, k1 = {};
    for (int kc = 0; kc < 8; ++kc) {
      v16h fa0 = frag_a_lds(valh + kc * 32, NREPR);
      v16h fa1 = frag_a_lds(valh + 16 * NREPR + kc * 32, NREPR);
      v16h fb = frag_pk(pkWk + (size_t)(kc * 13 + nt) * 512);
      k0 = wmma_f16(fa0, fb, k0);
      k1 = wmma_f16(fa1, fb, k1);
    }
    const int col = c0 + (lane & 15);
    const int sel = lane >> 4;
    const float bkv = (col < NQD) ? bk[col] : 0.f;
#pragma unroll
    for (int v = 0; v < 8; ++v) {
      int r = v + 8 * sel;
      keys[r * NQDP + col]        = (col < NQD) ? tanhf(k0[v] + bkv) : 0.f;
      keys[(16 + r) * NQDP + col] = (col < NQD) ? tanhf(k1[v] + bkv) : 0.f;
    }
  }
  __syncthreads();

  // additive-attention pooling
  if (tid < 32) {
    float acc = 0.f;
    if (tid < NT)
      for (int q = 0; q < NQD; ++q) acc += qvec[q] * keys[tid * NQDP + q];
    wl[tid] = acc * ATT_SCALE;
  }
  __syncthreads();
  if (tid < 32) {
    float m = -1e30f;
    for (int s = 0; s < NT; ++s) m = fmaxf(m, wl[s]);
    float sum = 0.f;
    for (int s = 0; s < NT; ++s) sum += __expf(wl[s] - m);
    wp[tid] = (tid < NT) ? __expf(wl[tid] - m) / sum : 0.f;
  }
  __syncthreads();
  {
    float acc = 0.f;
    for (int s = 0; s < NT; ++s) acc += wp[s] * val[s * NREPR + tid];
    outp[tid] = acc;
  }
}

// ===========================================================================
// Kernel 2: news-level user encoder.  One block per batch element.
// x = his_reprs[b]  [50 x 256] -> padded [64 x 256]
// ===========================================================================
__global__ void __launch_bounds__(256) nrms_user_encoder(
    const float* __restrict__ his_reprs, const _Float16* __restrict__ pkWq,
    const _Float16* __restrict__ pkWv, const _Float16* __restrict__ pkWk,
    const float* __restrict__ bk, const float* __restrict__ qvec,
    float* __restrict__ user_out) {
  extern __shared__ char smem[];
  float*    val  = (float*)smem;                  // [64][256]
  float*    sc   = val + 64 * NREPR;              // [8][64][64]; reused keys[64][208]
  _Float16* xs   = (_Float16*)(sc + 8 * 64 * 64); // [64][256]
  _Float16* valh = xs + 64 * NREPR;               // [64][256]
  _Float16* ach  = valh + 64 * NREPR;             // [8][64][32]
  _Float16* xvpk = ach + 8 * 64 * 32;             // [8][64][16]  (2 K-chunks)
  float*    wl   = (float*)(xvpk + 8 * 64 * 16);  // [64]
  float*    wp   = wl + 64;                       // [64]

  const int tid = threadIdx.x;
  const int w = tid >> 5, lane = tid & 31;
  const int b = blockIdx.x;

  for (int i = tid; i < 64 * NREPR; i += 256) {
    int s = i >> 8;
    xs[i] = (_Float16)((s < NHIS)
                ? his_reprs[(size_t)(b * NHIS + s) * NREPR + (i & 255)]
                : 0.f);
  }
  __syncthreads();

  float*    scw   = sc + w * 64 * 64;
  _Float16* achw  = ach + w * 64 * 32;
  _Float16* xvpkw = xvpk + w * 64 * 16;

  for (int hi = 0; hi < 2; ++hi) {
    const int h = w + hi * 8;
    const _Float16* Wqh = pkWq + (size_t)h * PK_WQN_HEAD;
    const _Float16* Wvh = pkWv + (size_t)h * PK_WVN_HEAD;
    __builtin_prefetch(Wqh, 0, 3);

    for (int i = lane; i < 64 * 64; i += 32) scw[i] = 0.f;

    for (int ec = 0; ec < 8; ++ec) {
      const int e0 = ec * 32;
      v8f acc[4][2];
#pragma unroll
      for (int mi = 0; mi < 4; ++mi) { acc[mi][0] = {}; acc[mi][1] = {}; }
      for (int kc = 0; kc < 8; ++kc) {
        v16h fb0 = frag_pk(Wqh + (size_t)(kc * 16 + 2 * ec) * 512);
        v16h fb1 = frag_pk(Wqh + (size_t)(kc * 16 + 2 * ec + 1) * 512);
#pragma unroll
        for (int mi = 0; mi < 4; ++mi) {
          v16h fa = frag_a_lds(xs + mi * 16 * NREPR + kc * 32, NREPR);
          acc[mi][0] = wmma_f16(fa, fb0, acc[mi][0]);
          acc[mi][1] = wmma_f16(fa, fb1, acc[mi][1]);
        }
      }
#pragma unroll
      for (int mi = 0; mi < 4; ++mi) {
        frag_c_store_f16(achw + mi * 16 * 32, 32, acc[mi][0]);
        frag_c_store_f16(achw + mi * 16 * 32 + 16, 32, acc[mi][1]);
      }
#pragma unroll
      for (int mi = 0; mi < 4; ++mi) {
        v16h fa = frag_a_lds(achw + mi * 16 * 32, 32);
#pragma unroll
        for (int ni = 0; ni < 4; ++ni) {
          v8f c = frag_c_load(scw + mi * 16 * 64 + ni * 16, 64);
          v16h fb = frag_bT_lds(xs + ni * 16 * NREPR + e0, NREPR);
          c = wmma_f16(fa, fb, c);
          frag_c_store(scw + mi * 16 * 64 + ni * 16, 64, c);
        }
      }
    }

    // row softmax over t<NHIS (in place, f32)
    for (int s = lane; s < 64; s += 32) {
      if (s < NHIS) {
        float m = -1e30f;
        for (int t = 0; t < NHIS; ++t) m = fmaxf(m, scw[s * 64 + t] * ATT_SCALE);
        float sum = 0.f;
        for (int t = 0; t < NHIS; ++t) sum += __expf(scw[s * 64 + t] * ATT_SCALE - m);
        float inv = 1.f / sum;
        for (int t = 0; t < NHIS; ++t)
          scw[s * 64 + t] = __expf(scw[s * 64 + t] * ATT_SCALE - m) * inv;
        for (int t = NHIS; t < 64; ++t) scw[s * 64 + t] = 0.f;
      } else {
        for (int t = 0; t < 64; ++t) scw[s * 64 + t] = 0.f;
      }
    }

    // xv = x @ Wv[h]   [64 x 16] -> packed-B layout (2 chunks)
    {
      v8f xv[4];
#pragma unroll
      for (int mi = 0; mi < 4; ++mi) xv[mi] = {};
      for (int kc = 0; kc < 8; ++kc) {
        v16h fb = frag_pk(Wvh + (size_t)kc * 512);
#pragma unroll
        for (int mi = 0; mi < 4; ++mi) {
          v16h fa = frag_a_lds(xs + mi * 16 * NREPR + kc * 32, NREPR);
          xv[mi] = wmma_f16(fa, fb, xv[mi]);
        }
      }
#pragma unroll
      for (int mi = 0; mi < 4; ++mi) store_xv_pk(xvpkw, mi, xv[mi]);
    }

    // val[:, h*16 : h*16+16] = attn @ xv   (K = 64 -> 2 chunks)
#pragma unroll
    for (int mi = 0; mi < 4; ++mi) {
      v8f a = {};
#pragma unroll
      for (int tc = 0; tc < 2; ++tc) {
        v16h fa = frag_a_f32(scw + mi * 16 * 64 + tc * 32, 64);
        v16h fb = frag_pk(xvpkw + tc * 512);
        a = wmma_f16(fa, fb, a);
      }
      frag_c_store(val + mi * 16 * NREPR + h * NVD, NREPR, a);
      frag_c_store_f16(valh + mi * 16 * NREPR + h * NVD, NREPR, a);
    }
  }
  __syncthreads();

  // keys reuse the (now dead) score region
  float* keys = sc;  // [64][208]
  for (int nt = w; nt < 13; nt += 8) {
    const int c0 = nt * 16;
    v8f k[4];
#pragma unroll
    for (int mi = 0; mi < 4; ++mi) k[mi] = {};
    for (int kc = 0; kc < 8; ++kc) {
      v16h fb = frag_pk(pkWk + (size_t)(kc * 13 + nt) * 512);
#pragma unroll
      for (int mi = 0; mi < 4; ++mi) {
        v16h fa = frag_a_lds(valh + mi * 16 * NREPR + kc * 32, NREPR);
        k[mi] = wmma_f16(fa, fb, k[mi]);
      }
    }
    const int col = c0 + (lane & 15);
    const int sel = lane >> 4;
    const float bkv = (col < NQD) ? bk[col] : 0.f;
#pragma unroll
    for (int mi = 0; mi < 4; ++mi)
#pragma unroll
      for (int v = 0; v < 8; ++v) {
        int r = mi * 16 + v + 8 * sel;
        keys[r * NQDP + col] = (col < NQD) ? tanhf(k[mi][v] + bkv) : 0.f;
      }
  }
  __syncthreads();

  if (tid < 64) {
    float acc = 0.f;
    if (tid < NHIS)
      for (int q = 0; q < NQD; ++q) acc += qvec[q] * keys[tid * NQDP + q];
    wl[tid] = acc * ATT_SCALE;
  }
  __syncthreads();
  if (tid < 64) {
    float m = -1e30f;
    for (int s = 0; s < NHIS; ++s) m = fmaxf(m, wl[s]);
    float sum = 0.f;
    for (int s = 0; s < NHIS; ++s) sum += __expf(wl[s] - m);
    wp[tid] = (tid < NHIS) ? __expf(wl[tid] - m) / sum : 0.f;
  }
  __syncthreads();
  {
    float acc = 0.f;
    for (int s = 0; s < NHIS; ++s) acc += wp[s] * val[s * NREPR + tid];
    user_out[(size_t)b * NREPR + tid] = acc;
  }
}

// ===========================================================================
// Kernel 3: dot-product scoring + log-softmax over candidates
// ===========================================================================
__global__ void __launch_bounds__(256) nrms_score(
    const float* __restrict__ cdd_reprs, const float* __restrict__ user_repr,
    float* __restrict__ out) {
  __shared__ float red[256];
  __shared__ float scb[NCDD];
  const int b = blockIdx.x, tid = threadIdx.x;
  const float u = user_repr[(size_t)b * NREPR + tid];
  for (int c = 0; c < NCDD; ++c) {
    red[tid] = cdd_reprs[((size_t)b * NCDD + c) * NREPR + tid] * u;
    __syncthreads();
    for (int off = 128; off > 0; off >>= 1) {
      if (tid < off) red[tid] += red[tid + off];
      __syncthreads();
    }
    if (tid == 0) scb[c] = red[0];
    __syncthreads();
  }
  if (tid == 0) {
    float m = -1e30f;
    for (int c = 0; c < NCDD; ++c) m = fmaxf(m, scb[c]);
    float sum = 0.f;
    for (int c = 0; c < NCDD; ++c) sum += __expf(scb[c] - m);
    float lse = m + __logf(sum);
    for (int c = 0; c < NCDD; ++c) out[(size_t)b * NCDD + c] = scb[c] - lse;
  }
}

// ===========================================================================
extern "C" void kernel_launch(void* const* d_in, const int* in_sizes, int n_in,
                              void* d_out, int out_size, void* d_ws, size_t ws_size,
                              hipStream_t stream) {
  const int*   cand    = (const int*)d_in[0];
  const int*   clicked = (const int*)d_in[1];
  const float* emb     = (const float*)d_in[2];
  const float* Wq_w    = (const float*)d_in[3];
  const float* Wv_w    = (const float*)d_in[4];
  const float* Wk_w    = (const float*)d_in[5];
  const float* bk_w    = (const float*)d_in[6];
  const float* q_w     = (const float*)d_in[7];
  const float* Wq_n    = (const float*)d_in[8];
  const float* Wv_n    = (const float*)d_in[9];
  const float* Wk_n    = (const float*)d_in[10];
  const float* bk_n    = (const float*)d_in[11];
  const float* q_n     = (const float*)d_in[12];
  float* out = (float*)d_out;

  float* cdd_reprs = (float*)d_ws;                              // [B,CDD,REPR]
  float* his_reprs = cdd_reprs + (size_t)NB * NCDD * NREPR;     // [B,HIS,REPR]
  float* user_repr = his_reprs + (size_t)NB * NHIS * NREPR;     // [B,REPR]

  _Float16* pkWqW = (_Float16*)(user_repr + (size_t)NB * NREPR);
  _Float16* pkWvW = pkWqW + (size_t)NH * PK_WQW_HEAD;
  _Float16* pkWkW = pkWvW + (size_t)NH * PK_WVW_HEAD;
  _Float16* pkWqN = pkWkW + PK_WKW_SZ;
  _Float16* pkWvN = pkWqN + (size_t)NH * PK_WQN_HEAD;
  _Float16* pkWkN = pkWvN + (size_t)NH * PK_WVN_HEAD;

  nrms_pack_b<<<1024, 256, 0, stream>>>(Wq_w, pkWqW, ND, ND, 10, 20, NH);
  nrms_pack_b<<<128, 256, 0, stream>>>(Wv_w, pkWvW, ND, NVD, 10, 1, NH);
  nrms_pack_b<<<64, 256, 0, stream>>>(Wk_w, pkWkW, NREPR, NQD, 8, 13, 1);
  nrms_pack_b<<<1024, 256, 0, stream>>>(Wq_n, pkWqN, NREPR, NREPR, 8, 16, NH);
  nrms_pack_b<<<64, 256, 0, stream>>>(Wv_n, pkWvN, NREPR, NVD, 8, 1, NH);
  nrms_pack_b<<<64, 256, 0, stream>>>(Wk_n, pkWkN, NREPR, NQD, 8, 13, 1);

  constexpr size_t SMEM1 =
      (32 * NREPR + 32 * NQDP + 8 * 32 * 32) * sizeof(float) +
      (32 * NDP + 32 * NREPR + 8 * 32 * 32 + 8 * 32 * 16) * sizeof(_Float16) +
      3 * 32 * sizeof(float);
  constexpr size_t SMEM2 =
      (64 * NREPR + 8 * 64 * 64) * sizeof(float) +
      (64 * NREPR + 64 * NREPR + 8 * 64 * 32 + 8 * 64 * 16) * sizeof(_Float16) +
      2 * 64 * sizeof(float);

  (void)hipFuncSetAttribute((const void*)nrms_word_encoder,
                            hipFuncAttributeMaxDynamicSharedMemorySize, (int)SMEM1);
  (void)hipFuncSetAttribute((const void*)nrms_user_encoder,
                            hipFuncAttributeMaxDynamicSharedMemorySize, (int)SMEM2);

  nrms_word_encoder<<<NB * (NCDD + NHIS), 256, SMEM1, stream>>>(
      cand, clicked, emb, pkWqW, pkWvW, pkWkW, bk_w, q_w, cdd_reprs, his_reprs);
  nrms_user_encoder<<<NB, 256, SMEM2, stream>>>(his_reprs, pkWqN, pkWvN, pkWkN,
                                                bk_n, q_n, user_repr);
  nrms_score<<<NB, 256, 0, stream>>>(cdd_reprs, user_repr, out);
}
